// RNNDecoderP_59468117180435
// MI455X (gfx1250) — compile-verified, matrix operand order checked by reading, not run on hardware
//
#include <hip/hip_runtime.h>
#include <hip/hip_bf16.h>

typedef __attribute__((ext_vector_type(16))) _Float16 v16h;
typedef __attribute__((ext_vector_type(8)))  _Float16 v8h;
typedef __attribute__((ext_vector_type(2)))  __fp16   pkh2;  // cvt_pkrtz result type
typedef __attribute__((ext_vector_type(8)))  float    v8f;

#define BB   256
#define TT   2048
#define DD   64
#define LBANDS 2
#define DBW  32      // band embedding dims
#define NL   2       // GRU layers
#define GG   96      // 3*DBW gate width
#define ROWPU 20     // LDS row pitch in pkh2 units (40 halves = 80B, 16B-aligned)

// Storage-slot index for A/B operands of v_wmma_*_16x16x32_f16:
// lanes 0-15 read contiguous slots {0..7, 16..23}; lanes 16-31 {8..15, 24..31}.
__device__ __forceinline__ int qof(int lane, int i) {
  return (i < 8 ? i : i + 8) + ((lane >> 4) << 3);
}
// Actual k held by slot q under the tile-interleaved LDS layout
// (position q = 2*(k%16) + k/16). A and B use the same permutation, so the
// WMMA reduction over k is unchanged.
__device__ __forceinline__ int kof(int lane, int i) {
  const int q = qof(lane, i);
  return ((q & 1) << 4) | (q >> 1);
}

__device__ __forceinline__ v8f wmma_f16(v16h a, v16h b, v8f c) {
  return __builtin_amdgcn_wmma_f32_16x16x32_f16(false, a, false, b, (short)0, c,
                                                false, false);
}

// fast sigmoid: v_exp_f32 + v_rcp_f32, no IEEE division
__device__ __forceinline__ float sigm(float x) {
  return __builtin_amdgcn_rcpf(1.0f + __builtin_amdgcn_exp2f(-1.44269504f * x));
}

// native V_TANH_F32 on gfx1250 when the builtin exists; branch-free fallback
#if defined(__has_builtin)
#if __has_builtin(__builtin_amdgcn_tanhf)
#define HAVE_AMDGCN_TANH 1
#endif
#endif
__device__ __forceinline__ float fast_tanh(float x) {
#ifdef HAVE_AMDGCN_TANH
  return __builtin_amdgcn_tanhf(x);
#else
  float ax = __builtin_fabsf(x);
  float e  = __builtin_amdgcn_exp2f(-2.88539008f * ax);   // e^(-2|x|)
  float t  = (1.0f - e) * __builtin_amdgcn_rcpf(1.0f + e);
  return __builtin_copysignf(t, x);
#endif
}

// compiler-level fence; LDS is in-order per wave (DScnt), wave owns its region
__device__ __forceinline__ void wave_lds_sync() {
  __asm__ __volatile__("" ::: "memory");
  __builtin_amdgcn_wave_barrier();
  __asm__ __volatile__("" ::: "memory");
}

// ---------------------------------------------------------------------------
// Kernel 1: stable compaction per (band,row) into a *transposed* layout
//   p_dtimeT[(kb*16+tile)][t][row%16]  (16 floats per step -> one 64B line)
// Permuted mask == (t < count).
// ---------------------------------------------------------------------------
__global__ void compact_kernel(const int* __restrict__ band_ids,
                               const float* __restrict__ dtime,
                               float* __restrict__ p_dtimeT,
                               int* __restrict__ cnt) {
  int idx = blockIdx.x * blockDim.x + threadIdx.x;
  if (idx >= LBANDS * BB) return;
  int kb = idx / BB, b = idx % BB;
  int tile = b >> 4, row = b & 15;
  const int*   bi = band_ids + (size_t)b * TT;
  const float* dt = dtime    + (size_t)b * TT;
  float* outp = p_dtimeT + ((size_t)(kb * 16 + tile)) * TT * 16 + row;
  int front = 0;
  for (int t = 0; t < TT; ++t)
    if (bi[t] == kb) outp[(size_t)(front++) * 16] = dt[t];
  cnt[kb * BB + b] = front;
  for (int t = 0; t < TT; ++t)
    if (bi[t] != kb) outp[(size_t)(front++) * 16] = dt[t];
}

// ---------------------------------------------------------------------------
// Kernel 2: rank-1 layer-0 precompute:
//   A0[b,g] = (z_last[b]·projW[:64] + proj_b)·Wih0[g,:] + bih0[g]
//   c0[g]   = projW[64,:]·Wih0[g,:]
// ---------------------------------------------------------------------------
__global__ void precompute_kernel(const float* __restrict__ z_last,
                                  const float* __restrict__ projW,
                                  const float* __restrict__ projb,
                                  const float* __restrict__ Wih,
                                  const float* __restrict__ bih,
                                  float* __restrict__ A0,
                                  float* __restrict__ c0) {
  int kb = blockIdx.x / BB, b = blockIdx.x % BB;
  __shared__ float base[DBW];
  int tid = threadIdx.x;
  if (tid < DBW) {
    float s = projb[kb * DBW + tid];
    const float* z = z_last + (size_t)b * DD;
    for (int d = 0; d < DD; ++d)
      s += z[d] * projW[((size_t)kb * (DD + 1) + d) * DBW + tid];
    base[tid] = s;
  }
  __syncthreads();
  {
    int g = tid;  // 0..95
    const float* w = Wih + (((size_t)kb * NL + 0) * GG + g) * DBW;
    float s = bih[(kb * NL + 0) * GG + g];
    for (int k = 0; k < DBW; ++k) s += base[k] * w[k];
    A0[((size_t)kb * BB + b) * GG + g] = s;
  }
  if (b == 0) {
    int g = tid;
    const float* w  = Wih + (((size_t)kb * NL + 0) * GG + g) * DBW;
    const float* wl = projW + ((size_t)kb * (DD + 1) + DD) * DBW;
    float s = 0.0f;
    for (int k = 0; k < DBW; ++k) s += wl[k] * w[k];
    c0[kb * GG + g] = s;
  }
}

// ---------------------------------------------------------------------------
// Kernel 3: fused 2-layer GRU + MLP head. One wave per (band, 16-row tile).
// Weights resident in VGPRs as WMMA B operands; biases/A0 folded into WMMA C.
// Serial loop (scalar-controlled via readfirstlane) runs only to
// tmax = max(count) over the tile; constant tail is a store-only fill loop.
// ---------------------------------------------------------------------------
__global__ __launch_bounds__(32) void gru_fused_kernel(
    const float* __restrict__ p_dtimeT, const int* __restrict__ cnt,
    const float* __restrict__ A0, const float* __restrict__ c0,
    const float* __restrict__ Wih, const float* __restrict__ Whh,
    const float* __restrict__ bih, const float* __restrict__ bhh,
    const float* __restrict__ mW1, const float* __restrict__ mb1,
    const float* __restrict__ mW2, const float* __restrict__ mb2,
    float* __restrict__ out) {
  __shared__ __align__(16) pkh2 st0[16 * ROWPU];
  __shared__ __align__(16) pkh2 st1[16 * ROWPU];
  const int lane = threadIdx.x;
  const int gw   = blockIdx.x;            // 0..31
  const int kb   = gw >> 4;               // band
  const int m0   = (gw & 15) * 16;        // batch tile base row
  const int col  = lane & 15;             // N-column within a WMMA tile
  const int hi   = lane >> 4;             // row-half select for C/D layout

  // ---- resident WMMA B operands ----
  v16h Bh0[6], Bx1[6], Bh1[6], Bm[2];
  float c0c[6];
  v8f Crz0[4];   // A0 rows + bhh0 (layer0 r/z accumulate base)
  v8f a0n[2];    // A0 rows for n-gate x-part (tiles 4,5)
  v8f Cn0h[2];   // splat(bhh0) for layer0 n-gate gh wmma
  v8f Crz1[4];   // splat(bih1+bhh1) for layer1 r/z chain base
  v8f Cn1x[2];   // splat(bih1) tiles 4,5
  v8f Cn1h[2];   // splat(bhh1) tiles 4,5
  v8f Cmlp[2];   // splat(mb1)
  for (int n = 0; n < 6; ++n) {
    const int g = 16 * n + col;
    const float* wh0 = Whh + (((size_t)kb * NL + 0) * GG + g) * DBW;
    const float* wx1 = Wih + (((size_t)kb * NL + 1) * GG + g) * DBW;
    const float* wh1 = Whh + (((size_t)kb * NL + 1) * GG + g) * DBW;
    for (int i = 0; i < 16; ++i) {
      const int k = kof(lane, i);
      Bh0[n][i] = (_Float16)wh0[k];
      Bx1[n][i] = (_Float16)wx1[k];
      Bh1[n][i] = (_Float16)wh1[k];
    }
    const float b0h = bhh[(kb * NL + 0) * GG + g];
    const float b1i = bih[(kb * NL + 1) * GG + g];
    const float b1h = bhh[(kb * NL + 1) * GG + g];
    c0c[n] = c0[kb * GG + g];
    v8f a0row;
    for (int j = 0; j < 8; ++j)
      a0row[j] = A0[((size_t)kb * BB + m0 + j + 8 * hi) * GG + g];
    if (n < 4) {
      for (int j = 0; j < 8; ++j) Crz0[n][j] = a0row[j] + b0h;
      for (int j = 0; j < 8; ++j) Crz1[n][j] = b1i + b1h;
    } else {
      a0n[n - 4] = a0row;
      for (int j = 0; j < 8; ++j) Cn0h[n - 4][j] = b0h;
      for (int j = 0; j < 8; ++j) Cn1x[n - 4][j] = b1i;
      for (int j = 0; j < 8; ++j) Cn1h[n - 4][j] = b1h;
    }
  }
  float w2c[2];
  for (int p = 0; p < 2; ++p) {
    const int n2 = 16 * p + col;
    for (int i = 0; i < 16; ++i)
      Bm[p][i] = (_Float16)mW1[((size_t)kb * DBW + kof(lane, i)) * DBW + n2];
    const float mb = mb1[kb * DBW + n2];
    for (int j = 0; j < 8; ++j) Cmlp[p][j] = mb;
    w2c[p] = mW2[kb * DBW + n2];
  }
  const float mb2v = mb2[kb];

  // per-row valid-step counts (mask == t < cnt); tmax = max over tile (SGPR)
  int crow = (lane < 16) ? cnt[kb * BB + m0 + lane] : 0;
  int cj[8];
  for (int j = 0; j < 8; ++j) cj[j] = __shfl(crow, j + 8 * hi, 32);
  int lm = 0;
  for (int j = 0; j < 8; ++j) lm = max(lm, cj[j]);
  lm = max(lm, __shfl_xor(lm, 16, 32));
  const int tmax = __builtin_amdgcn_readfirstlane(lm);  // scalar loop bound

  const float* dtt = p_dtimeT + (size_t)gw * TT * 16 + 8 * hi;
  float* yp = out + (size_t)kb * BB * TT + (size_t)m0 * TT;

  // A-operand read base (row = col) in halves within the staging buffer
  const int rbase = col * (2 * ROWPU) + hi * 8;

  v16h h0A = {};
  v16h h1A = {};
  v8f hD0[2]; hD0[0] = v8f{}; hD0[1] = v8f{};
  v8f hD1[2]; hD1[0] = v8f{}; hD1[1] = v8f{};

  // MLP head: relu(h1 @ mW1 + mb1) @ mW2 ; row-sums land in all lanes
  auto mlp_eval = [&](const v16h& a, float* part) {
    v8f m0v = wmma_f16(a, Bm[0], Cmlp[0]);
    v8f m1v = wmma_f16(a, Bm[1], Cmlp[1]);
    for (int j = 0; j < 8; ++j)
      part[j] = fmaxf(m0v[j], 0.0f) * w2c[0] + fmaxf(m1v[j], 0.0f) * w2c[1];
    for (int mk = 1; mk < 16; mk <<= 1)
      for (int j = 0; j < 8; ++j)
        part[j] += __shfl_xor(part[j], mk, 32);
  };

  // software-pipelined per-row dtime (16 floats per step = one 64B line)
  float4 nA = *(const float4*)(dtt);
  float4 nB = *(const float4*)(dtt + 4);

  for (int t = 0; t < tmax; ++t) {
    const float4 dA = nA, dB = nB;
    const int tn = (t + 1 < TT) ? t + 1 : t;
    nA = *(const float4*)(dtt + (size_t)tn * 16);
    nB = *(const float4*)(dtt + (size_t)tn * 16 + 4);
    const float dtj[8] = {dA.x, dA.y, dA.z, dA.w, dB.x, dB.y, dB.z, dB.w};

    // -------- layer 0: gates = (A0 + dt*c0 + bih0) + h0@Whh0^T + bhh0 ------
    v8f grz0[4], ghn0[2];
    for (int n = 0; n < 4; ++n) grz0[n] = wmma_f16(h0A, Bh0[n], Crz0[n]);
    ghn0[0] = wmma_f16(h0A, Bh0[4], Cn0h[0]);
    ghn0[1] = wmma_f16(h0A, Bh0[5], Cn0h[1]);
    {
      v8f hn0, hn1;
      for (int j = 0; j < 8; ++j) {
        const float r0  = sigm(grz0[0][j] + dtj[j] * c0c[0]);
        const float z0  = sigm(grz0[2][j] + dtj[j] * c0c[2]);
        const float nn0 = fast_tanh(a0n[0][j] + dtj[j] * c0c[4] + r0 * ghn0[0][j]);
        const float r1  = sigm(grz0[1][j] + dtj[j] * c0c[1]);
        const float z1  = sigm(grz0[3][j] + dtj[j] * c0c[3]);
        const float nn1 = fast_tanh(a0n[1][j] + dtj[j] * c0c[5] + r1 * ghn0[1][j]);
        const float hp0 = hD0[0][j], hp1 = hD0[1][j];
        const bool  live = (t < cj[j]);
        hn0[j] = live ? (nn0 + z0 * (hp0 - nn0)) : hp0;
        hn1[j] = live ? (nn1 + z1 * (hp1 - nn1)) : hp1;
      }
      hD0[0] = hn0; hD0[1] = hn1;
      // packed f16 store: both N-tiles of row j in one ds_store_b32
      for (int j = 0; j < 8; ++j)
        st0[(j + 8 * hi) * ROWPU + col] = __builtin_amdgcn_cvt_pkrtz(hn0[j], hn1[j]);
    }
    wave_lds_sync();
    {
      const _Float16* sp = (const _Float16*)(const void*)st0 + rbase;
      const v8h lo  = *(const v8h*)(sp);
      const v8h hi8 = *(const v8h*)(sp + 16);
      for (int i = 0; i < 8; ++i) { h0A[i] = lo[i]; h0A[i + 8] = hi8[i]; }
    }

    // -------- layer 1: r/z chained (gx+gh in one accumulation chain) -------
    v8f s1[4];
    for (int n = 0; n < 4; ++n) s1[n] = wmma_f16(h0A, Bx1[n], Crz1[n]);
    for (int n = 0; n < 4; ++n) s1[n] = wmma_f16(h1A, Bh1[n], s1[n]);
    v8f gxn[2], ghn1[2];
    gxn[0]  = wmma_f16(h0A, Bx1[4], Cn1x[0]);
    gxn[1]  = wmma_f16(h0A, Bx1[5], Cn1x[1]);
    ghn1[0] = wmma_f16(h1A, Bh1[4], Cn1h[0]);
    ghn1[1] = wmma_f16(h1A, Bh1[5], Cn1h[1]);
    {
      v8f hn0, hn1;
      for (int j = 0; j < 8; ++j) {
        const float r0  = sigm(s1[0][j]);
        const float z0  = sigm(s1[2][j]);
        const float nn0 = fast_tanh(gxn[0][j] + r0 * ghn1[0][j]);
        const float r1  = sigm(s1[1][j]);
        const float z1  = sigm(s1[3][j]);
        const float nn1 = fast_tanh(gxn[1][j] + r1 * ghn1[1][j]);
        const float hp0 = hD1[0][j], hp1 = hD1[1][j];
        const bool  live = (t < cj[j]);
        hn0[j] = live ? (nn0 + z0 * (hp0 - nn0)) : hp0;
        hn1[j] = live ? (nn1 + z1 * (hp1 - nn1)) : hp1;
      }
      hD1[0] = hn0; hD1[1] = hn1;
      for (int j = 0; j < 8; ++j)
        st1[(j + 8 * hi) * ROWPU + col] = __builtin_amdgcn_cvt_pkrtz(hn0[j], hn1[j]);
    }
    wave_lds_sync();
    {
      const _Float16* sp = (const _Float16*)(const void*)st1 + rbase;
      const v8h lo  = *(const v8h*)(sp);
      const v8h hi8 = *(const v8h*)(sp + 16);
      for (int i = 0; i < 8; ++i) { h1A[i] = lo[i]; h1A[i + 8] = hi8[i]; }
    }

    // -------- MLP head for this step --------
    float part[8];
    mlp_eval(h1A, part);
    if (col == 0) {  // lane 0 -> rows 0..7, lane 16 -> rows 8..15
      for (int j = 0; j < 8; ++j)
        yp[(size_t)(j + 8 * hi) * TT + t] = part[j] + mb2v;
    }
  }

  // -------- frozen tail: y constant = MLP(final h1); parallel fill --------
  float part[8];
  mlp_eval(h1A, part);   // also correct when tmax == 0 (h == 0)
  for (int j = 0; j < 8; ++j) {
    const float yv = part[j] + mb2v;
    float* yrow = yp + (size_t)(j + 8 * hi) * TT;
    for (int t = tmax + col; t < TT; t += 16)
      yrow[t] = yv;      // 16 lanes per half cover 64B of a row per iteration
  }
}

// ---------------------------------------------------------------------------
extern "C" void kernel_launch(void* const* d_in, const int* in_sizes, int n_in,
                              void* d_out, int out_size, void* d_ws, size_t ws_size,
                              hipStream_t stream) {
  const int*   band_ids = (const int*)d_in[0];
  const float* dtime    = (const float*)d_in[1];
  const float* z_last   = (const float*)d_in[2];
  const float* projW    = (const float*)d_in[3];
  const float* projb    = (const float*)d_in[4];
  const float* Wih      = (const float*)d_in[5];
  const float* Whh      = (const float*)d_in[6];
  const float* bih      = (const float*)d_in[7];
  const float* bhh      = (const float*)d_in[8];
  const float* mW1      = (const float*)d_in[9];
  const float* mb1      = (const float*)d_in[10];
  const float* mW2      = (const float*)d_in[11];
  const float* mb2      = (const float*)d_in[12];
  float* outp = (float*)d_out;

  // workspace carve-up (~4.4 MB total)
  float* p_dtimeT = (float*)d_ws;
  int*   cnt      = (int*)(p_dtimeT + (size_t)LBANDS * BB * TT);
  float* A0       = (float*)(cnt + LBANDS * BB);
  float* c0       = A0 + (size_t)LBANDS * BB * GG;

  compact_kernel<<<(LBANDS * BB + 255) / 256, 256, 0, stream>>>(
      band_ids, dtime, p_dtimeT, cnt);
  precompute_kernel<<<LBANDS * BB, GG, 0, stream>>>(
      z_last, projW, projb, Wih, bih, A0, c0);
  gru_fused_kernel<<<32, 32, 0, stream>>>(
      p_dtimeT, cnt, A0, c0, Wih, Whh, bih, bhh, mW1, mb1, mW2, mb2, outp);
}
